// LSA_39333310497193
// MI455X (gfx1250) — compile-verified, matrix operand order checked by reading, not run on hardware
//
#include <hip/hip_runtime.h>

typedef __attribute__((ext_vector_type(16))) _Float16 v16h;
typedef __attribute__((ext_vector_type(8)))  float    v8f;

#define WMMA16(a, b, c) __builtin_amdgcn_wmma_f32_16x16x32_f16( \
    false, (a), false, (b), (short)0, (c), false, false)

static constexpr int kB    = 8;
static constexpr int kN    = 1024;
static constexpr int kDim  = 512;
static constexpr int kH    = 8;
static constexpr int kDh   = 64;
static constexpr int kInner= 512;   // kH * kDh
static constexpr int kQKV  = 1536;  // 3 * kInner
static constexpr int kTok  = kB * kN;  // 8192

// ---------------------------------------------------------------------------
// Fragment gather: works for A fragments (row i0+lane%16, contiguous K) and,
// because the B layout mirrors A (lane%16 = N column, same K mapping), also
// for B fragments from an N-major (transposed) matrix.
// Per-lane mapping (ISA 7.12.2, 16-bit A 16x32):
//   element e -> K = k0 + (e<8?0:16) + (lane>>4)*8 + (e&7)
// Compiles to two global_load_b128 per fragment.
// ---------------------------------------------------------------------------
__device__ __forceinline__ v16h load_frag(const _Float16* p, int ld, int i0, int k0) {
    const int lane = threadIdx.x & 31;
    const _Float16* base = p + (size_t)(i0 + (lane & 15)) * ld + k0 + ((lane >> 4) << 3);
    v16h r;
#pragma unroll
    for (int e = 0; e < 8; ++e) { r[e] = base[e]; r[e + 8] = base[e + 16]; }
    return r;
}

__device__ __forceinline__ float rowmax16(float v) {
#pragma unroll
    for (int m = 1; m < 16; m <<= 1) v = fmaxf(v, __shfl_xor(v, m, 32));
    return v;
}
__device__ __forceinline__ float rowsum16(float v) {
#pragma unroll
    for (int m = 1; m < 16; m <<= 1) v += __shfl_xor(v, m, 32);
    return v;
}

// ---------------------------------------------------------------------------
// f32 -> f16 elementwise convert
// ---------------------------------------------------------------------------
__global__ void cvt_f16_kernel(const float* __restrict__ in, _Float16* __restrict__ out, int n) {
    int i = blockIdx.x * blockDim.x + threadIdx.x;
    if (i < n) out[i] = (_Float16)in[i];
}

// transpose+convert: in (rows x cols) row-major f32 -> out (cols x rows) f16
__global__ void tr_f16_kernel(const float* __restrict__ in, _Float16* __restrict__ out,
                              int rows, int cols) {
    int i = blockIdx.x * blockDim.x + threadIdx.x;
    int n = rows * cols;
    if (i >= n) return;
    int r = i / cols, c = i % cols;
    out[(size_t)c * rows + r] = (_Float16)in[i];
}

// ---------------------------------------------------------------------------
// QKV projection: xh (8192 x 512) @ wqkv_t^T  (wqkv_t is 1536 x 512, N-major)
// Scatter results into q (B,H,N,D), k (B,H,N,D), v transposed (B,H,D,N), f16.
// Grid: (32, 24), block 256 (8 waves); wave = 32 rows x 64 cols of output.
// Software-pipelined (prefetch next K-chunk), unroll 2 for mov-free
// double-buffer ping-pong.
// ---------------------------------------------------------------------------
__global__ void qkv_gemm_kernel(const _Float16* __restrict__ xh,
                                const _Float16* __restrict__ wqkv_t,
                                _Float16* __restrict__ qh,
                                _Float16* __restrict__ kh,
                                _Float16* __restrict__ vt) {
    const int wave = threadIdx.x >> 5;
    const int lane = threadIdx.x & 31;
    const int r0 = blockIdx.x * 256 + wave * 32;   // token row
    const int c0 = blockIdx.y * 64;                // qkv column

    v8f acc[2][4] = {};
    v16h a[2], b[4];
    a[0] = load_frag(xh, kDim, r0,      0);
    a[1] = load_frag(xh, kDim, r0 + 16, 0);
#pragma unroll
    for (int t = 0; t < 4; ++t) b[t] = load_frag(wqkv_t, kDim, c0 + t * 16, 0);

#pragma unroll 2
    for (int k0 = 0; k0 < kDim; k0 += 32) {
        const int kn = (k0 + 32) & (kDim - 1);     // wrap: dead prefetch on last iter
        v16h an[2], bn[4];
        an[0] = load_frag(xh, kDim, r0,      kn);
        an[1] = load_frag(xh, kDim, r0 + 16, kn);
#pragma unroll
        for (int t = 0; t < 4; ++t) bn[t] = load_frag(wqkv_t, kDim, c0 + t * 16, kn);

#pragma unroll
        for (int t = 0; t < 4; ++t) {
            acc[0][t] = WMMA16(a[0], b[t], acc[0][t]);
            acc[1][t] = WMMA16(a[1], b[t], acc[1][t]);
        }
        a[0] = an[0]; a[1] = an[1];
#pragma unroll
        for (int t = 0; t < 4; ++t) b[t] = bn[t];
    }

    const int nloc  = lane & 15;
    const int mbase = (lane >> 4) * 8;
#pragma unroll
    for (int half = 0; half < 2; ++half) {
#pragma unroll
        for (int t = 0; t < 4; ++t) {
            int c     = c0 + t * 16 + nloc;
            int which = c / kInner;          // 0=q 1=k 2=v
            int inner = c % kInner;
            int h     = inner / kDh;
            int d     = inner % kDh;
#pragma unroll
            for (int j = 0; j < 8; ++j) {
                int row = r0 + half * 16 + mbase + j;   // global token
                int bb  = row >> 10;
                int n   = row & (kN - 1);
                _Float16 hv = (_Float16)acc[half][t][j];
                size_t hb = (size_t)(bb * kH + h);
                if (which == 0)      qh[(hb * kN + n) * kDh + d] = hv;
                else if (which == 1) kh[(hb * kN + n) * kDh + d] = hv;
                else                 vt[(hb * kDh + d) * kN + n] = hv;
            }
        }
    }
}

// ---------------------------------------------------------------------------
// Flash attention with diagonal mask. Grid: (8, 64): y = b*H+h, x = row block.
// Each of 8 waves owns 16 query rows; streams K/V in 32-column steps with
// K-fragment prefetch; softmax VALU work hides V/K load latency.
// ---------------------------------------------------------------------------
__global__ void attn_kernel(const _Float16* __restrict__ qh,
                            const _Float16* __restrict__ kh,
                            const _Float16* __restrict__ vt,
                            const float* __restrict__ temperature,
                            _Float16* __restrict__ ao) {
    const int wave = threadIdx.x >> 5;
    const int lane = threadIdx.x & 31;
    const int bh   = blockIdx.y;
    const int bb   = bh >> 3;
    const int h    = bh & 7;
    const int q0   = blockIdx.x * 128 + wave * 16;   // query row within (b,h)

    const float scale = __expf(temperature[0]);

    const _Float16* qp = qh + (size_t)bh * kN * kDh;
    const _Float16* kp = kh + (size_t)bh * kN * kDh;
    const _Float16* vp = vt + (size_t)bh * kDh * kN;

    __shared__ _Float16 pbuf[8][16 * 32];   // per-wave P staging (1 KB each)

    const v16h qa0 = load_frag(qp, kDh, q0, 0);
    const v16h qa1 = load_frag(qp, kDh, q0, 32);

    const int nloc  = lane & 15;
    const int mbase = (lane >> 4) * 8;

    float mrow[8], lrow[8];
    v8f o[4] = {};
#pragma unroll
    for (int j = 0; j < 8; ++j) { mrow[j] = -3.0e38f; lrow[j] = 0.0f; }

    // prologue: K fragments for first j-step
    v16h kb[4];
    kb[0] = load_frag(kp, kDh, 0,  0);
    kb[1] = load_frag(kp, kDh, 0,  32);
    kb[2] = load_frag(kp, kDh, 16, 0);
    kb[3] = load_frag(kp, kDh, 16, 32);

#pragma unroll 2
    for (int j0 = 0; j0 < kN; j0 += 32) {
        // ---- S = Q K^T for two 16-wide column tiles, K-dim = 64 ----
        v8f s0 = v8f{}, s1 = v8f{};
        s0 = WMMA16(qa0, kb[0], s0);
        s0 = WMMA16(qa1, kb[1], s0);
        s1 = WMMA16(qa0, kb[2], s1);
        s1 = WMMA16(qa1, kb[3], s1);

        // ---- prefetch: V fragments for THIS step (needed after softmax),
        //      then K fragments for NEXT step (needed next iteration) ----
        v16h vb[4], kbn[4];
#pragma unroll
        for (int t = 0; t < 4; ++t) vb[t] = load_frag(vp, kN, t * 16, j0);
        const int jn = (j0 + 32) & (kN - 1);       // wrap: dead prefetch on last
        kbn[0] = load_frag(kp, kDh, jn,      0);
        kbn[1] = load_frag(kp, kDh, jn,      32);
        kbn[2] = load_frag(kp, kDh, jn + 16, 0);
        kbn[3] = load_frag(kp, kDh, jn + 16, 32);

        // ---- scale, diagonal mask, online softmax update ----
#pragma unroll
        for (int j = 0; j < 8; ++j) {
            float a = s0[j] * scale;
            float b = s1[j] * scale;
            int grow = q0 + mbase + j;
            if (grow == j0 + nloc)      a = -3.402823466e38f;
            if (grow == j0 + 16 + nloc) b = -3.402823466e38f;
            float tmax = rowmax16(fmaxf(a, b));
            float newm = fmaxf(mrow[j], tmax);
            float corr = __expf(mrow[j] - newm);
            mrow[j] = newm;
            float p0 = __expf(a - newm);
            float p1 = __expf(b - newm);
            lrow[j] = lrow[j] * corr + rowsum16(p0 + p1);
#pragma unroll
            for (int t = 0; t < 4; ++t) o[t][j] *= corr;
            s0[j] = p0;
            s1[j] = p1;
        }

        // ---- C/D layout -> A layout via per-wave LDS round trip ----
        _Float16* pb = &pbuf[wave][0];
#pragma unroll
        for (int j = 0; j < 8; ++j) {
            int m = mbase + j;
            pb[m * 32 + nloc]      = (_Float16)s0[j];
            pb[m * 32 + 16 + nloc] = (_Float16)s1[j];
        }
        asm volatile("s_wait_dscnt 0" ::: "memory");
        v16h pf;
        {
            const _Float16* bp = pb + (lane & 15) * 32 + ((lane >> 4) << 3);
#pragma unroll
            for (int e = 0; e < 8; ++e) { pf[e] = bp[e]; pf[e + 8] = bp[e + 16]; }
        }
        asm volatile("" ::: "memory");

        // ---- O += P V   (V stored transposed: rows = d, contiguous in j) ----
#pragma unroll
        for (int t = 0; t < 4; ++t) o[t] = WMMA16(pf, vb[t], o[t]);

#pragma unroll
        for (int t = 0; t < 4; ++t) kb[t] = kbn[t];
    }

    // ---- normalize and emit (B, N, H*D) f16 for the output projection ----
#pragma unroll
    for (int t = 0; t < 4; ++t) {
#pragma unroll
        for (int j = 0; j < 8; ++j) {
            int row = q0 + mbase + j;
            int d   = t * 16 + nloc;
            float val = o[t][j] / lrow[j];
            ao[((size_t)(bb * kN + row)) * kInner + h * kDh + d] = (_Float16)val;
        }
    }
}

// ---------------------------------------------------------------------------
// Output projection: ao (8192 x 512) @ wout_t^T + b_out -> f32 out
// Grid: (32, 8), block 256; wave = 32 rows x 64 cols, software-pipelined.
// ---------------------------------------------------------------------------
__global__ void proj_kernel(const _Float16* __restrict__ ao,
                            const _Float16* __restrict__ wout_t,
                            const float* __restrict__ bout,
                            float* __restrict__ out) {
    const int wave = threadIdx.x >> 5;
    const int lane = threadIdx.x & 31;
    const int r0 = blockIdx.x * 256 + wave * 32;
    const int c0 = blockIdx.y * 64;

    v8f acc[2][4] = {};
    v16h a[2], b[4];
    a[0] = load_frag(ao, kInner, r0,      0);
    a[1] = load_frag(ao, kInner, r0 + 16, 0);
#pragma unroll
    for (int t = 0; t < 4; ++t) b[t] = load_frag(wout_t, kInner, c0 + t * 16, 0);

#pragma unroll 2
    for (int k0 = 0; k0 < kInner; k0 += 32) {
        const int kn = (k0 + 32) & (kInner - 1);
        v16h an[2], bn[4];
        an[0] = load_frag(ao, kInner, r0,      kn);
        an[1] = load_frag(ao, kInner, r0 + 16, kn);
#pragma unroll
        for (int t = 0; t < 4; ++t) bn[t] = load_frag(wout_t, kInner, c0 + t * 16, kn);

#pragma unroll
        for (int t = 0; t < 4; ++t) {
            acc[0][t] = WMMA16(a[0], b[t], acc[0][t]);
            acc[1][t] = WMMA16(a[1], b[t], acc[1][t]);
        }
        a[0] = an[0]; a[1] = an[1];
#pragma unroll
        for (int t = 0; t < 4; ++t) b[t] = bn[t];
    }

    const int nloc  = lane & 15;
    const int mbase = (lane >> 4) * 8;
#pragma unroll
    for (int half = 0; half < 2; ++half) {
#pragma unroll
        for (int t = 0; t < 4; ++t) {
            int c = c0 + t * 16 + nloc;
            float bias = bout[c];
#pragma unroll
            for (int j = 0; j < 8; ++j) {
                int row = r0 + half * 16 + mbase + j;
                out[(size_t)row * kDim + c] = acc[half][t][j] + bias;
            }
        }
    }
}

// ---------------------------------------------------------------------------
extern "C" void kernel_launch(void* const* d_in, const int* in_sizes, int n_in,
                              void* d_out, int out_size, void* d_ws, size_t ws_size,
                              hipStream_t stream) {
    const float* x    = (const float*)d_in[0];   // (8,1024,512)
    const float* wqkv = (const float*)d_in[1];   // (512,1536)
    const float* temp = (const float*)d_in[2];   // scalar
    const float* wout = (const float*)d_in[3];   // (512,512)
    const float* bout = (const float*)d_in[4];   // (512,)
    float* out = (float*)d_out;                  // (8,1024,512)

    // Workspace layout (f16 buffers), ~42 MB total
    char* ws = (char*)d_ws;
    _Float16* xh     = (_Float16*)(ws);                       // 8 MB
    _Float16* wqkv_t = (_Float16*)(ws + (8u   << 20));        // 1.5 MB
    _Float16* wout_t = (_Float16*)(ws + (9728u << 10));       // 0.5 MB  (9.5 MB off)
    _Float16* qh     = (_Float16*)(ws + (10u  << 20));        // 8 MB
    _Float16* kh     = (_Float16*)(ws + (18u  << 20));        // 8 MB
    _Float16* vt     = (_Float16*)(ws + (26u  << 20));        // 8 MB
    _Float16* ao     = (_Float16*)(ws + (34u  << 20));        // 8 MB

    const int nX = kTok * kDim;          // 4,194,304
    cvt_f16_kernel<<<(nX + 255) / 256, 256, 0, stream>>>(x, xh, nX);

    const int nWq = kDim * kQKV;         // 786,432
    tr_f16_kernel<<<(nWq + 255) / 256, 256, 0, stream>>>(wqkv, wqkv_t, kDim, kQKV);

    const int nWo = kInner * kDim;       // 262,144
    tr_f16_kernel<<<(nWo + 255) / 256, 256, 0, stream>>>(wout, wout_t, kInner, kDim);

    qkv_gemm_kernel<<<dim3(kTok / 256, kQKV / 64), 256, 0, stream>>>(xh, wqkv_t, qh, kh, vt);

    attn_kernel<<<dim3(kN / 128, kB * kH), 256, 0, stream>>>(qh, kh, vt, temp, ao);

    proj_kernel<<<dim3(kTok / 256, kDim / 64), 256, 0, stream>>>(ao, wout_t, bout, out);
}
